// CrossSpectrumAttention_44710609551531
// MI455X (gfx1250) — matrix operand order repaired and checked by reading, not compiled, not run on hardware
//
#include <hip/hip_runtime.h>
#include <hip/hip_bf16.h>

// ---------------------------------------------------------------------------
// CrossSpectrumAttention for MI455X (gfx1250, wave32, WMMA f16 16x16x32)
// ---------------------------------------------------------------------------

typedef _Float16 h16;
typedef __attribute__((ext_vector_type(16))) _Float16 v16h;
typedef __attribute__((ext_vector_type(8)))  _Float16 v8h;
typedef __attribute__((ext_vector_type(8)))  float    v8f;
typedef __attribute__((ext_vector_type(4)))  float    v4f;
typedef __attribute__((ext_vector_type(4)))  unsigned int v4u;
typedef __attribute__((ext_vector_type(8)))  unsigned short v8us;

union ABfrag { v16h v; v8h h[2]; };

__device__ __forceinline__ v8f wmma16(v16h a, v16h b, v8f c) {
    return __builtin_amdgcn_wmma_f32_16x16x32_f16(
        /*neg_a=*/false, a, /*neg_b=*/false, b,
        /*c_mod=*/(short)0, c, /*reuse_a=*/false, /*reuse_b=*/false);
}

__device__ __forceinline__ unsigned short f2h(float f) {
    union { _Float16 h; unsigned short u; } x;
    x.h = (_Float16)f;
    return x.u;
}

// Problem constants
constexpr int BATCH = 16;
constexpr int C_    = 256;
constexpr int HH    = 96;
constexpr int WW    = 96;
constexpr int HW    = HH * WW;       // 9216
constexpr int OUTC  = 64;
constexpr int CC    = C_ * C_;       // 65536

// Workspace layout (bytes)
constexpr size_t OFF_PE = 0;
constexpr size_t SZ_PE  = (size_t)C_ * HW * sizeof(float);       // 9,437,184
constexpr size_t OFF_WH = OFF_PE + SZ_PE;
constexpr size_t NWH    = (size_t)9 * CC + (size_t)OUTC * C_;    // 606,208 halfs
constexpr size_t SZ_WH  = NWH * 2;
constexpr size_t OFF_Q  = OFF_WH + SZ_WH;
constexpr size_t SZ_T   = (size_t)BATCH * C_ * HW * 2;           // 75,497,472
constexpr size_t OFF_K  = OFF_Q + SZ_T;
constexpr size_t OFF_V  = OFF_K + SZ_T;
constexpr size_t OFF_F  = OFF_V + SZ_T;

// ---------------------------------------------------------------------------
// Positional encoding: pe[o][n], n = h*96+w.
// pe[4m]   = sin(px*f) + sin(py*f),  pe[4m+1] = cos(px*f) + cos(py*f),
// f = exp(-2m * ln(1e4)/128), px = w/95, py = h/95; other channels = 0.
// ---------------------------------------------------------------------------
__global__ void pe_kernel(float* __restrict__ pe) {
    int idx = blockIdx.x * 256 + threadIdx.x;       // C_*HW = 9216*256 exactly
    int o = idx / HW, n = idx - o * HW;
    int h = n / WW, w = n - h * WW;
    float px = (float)w * (1.0f / 95.0f);
    float py = (float)h * (1.0f / 95.0f);
    float val = 0.0f;
    int m4 = o & 3;
    if (m4 < 2) {
        int m = o >> 2;
        float f = expf((float)(2 * m) * (-0.07195578415606394f)); // ln(1e4)/128
        val = (m4 == 0) ? (sinf(px * f) + sinf(py * f))
                        : (cosf(px * f) + cosf(py * f));
    }
    pe[idx] = val;
}

// fp32 -> f16 weight preconversion: [q_w(3*CC)][k_w(3*CC)][v_w(3*CC)][proj(64*256)]
__global__ void wcvt_kernel(const float* __restrict__ q, const float* __restrict__ k,
                            const float* __restrict__ v, const float* __restrict__ p,
                            unsigned short* __restrict__ out) {
    int i = blockIdx.x * 256 + threadIdx.x;         // 2368*256 == 606208 exactly
    const int S3 = 3 * CC;
    float f;
    if      (i <     S3) f = q[i];
    else if (i < 2 * S3) f = k[i - S3];
    else if (i < 3 * S3) f = v[i - 2 * S3];
    else                 f = p[i - 3 * S3];
    out[i] = f2h(f);
}

// ---------------------------------------------------------------------------
// Channel-mix GEMM:  D[n, o] = sum_c X[b,c,n] * W[o,c]   (WMMA f16, f32 acc)
// Block: 256 thr = 8 waves. Block tile: n=256 (8 waves x 32), o=64.
// Wave tile 32(n) x 64(o): 2 A-frags x 4 B-frags -> 8 WMMA per 12 ds_load_b128
// (1.5 loads/WMMA). Xt staging stores are lane-rotated to avoid 16-way bank
// conflicts (16*pitch bytes is 0 mod 256B).
// MODE 0: Y = f16( scale[n]*acc + bias[o] + pe[o,n] )   (q / k)
// MODE 1: Y = f16( scale[n]*acc + bias[o] )             (v)
// MODE 2: Y = f32( acc + bias[o] )                      (projection)
// ---------------------------------------------------------------------------
constexpr int XT_P = 40;    // Xt pitch in halfs (32 + 8 pad, 16B-aligned)
constexpr int WL_P = 264;   // Wlds pitch in halfs (256 + 8 pad)

template<bool IN_F16, int MODE>
__global__ void __launch_bounds__(256)
conv_gemm(const void* __restrict__ Xv, const unsigned short* __restrict__ Wh,
          const float* __restrict__ bias, const float* __restrict__ scale,
          const float* __restrict__ pe, void* __restrict__ Yv)
{
    __shared__ unsigned short Wlds[64 * WL_P];    // 33,792 B
    __shared__ unsigned short Xt[256 * XT_P];     // 20,480 B

    const int t     = threadIdx.x;
    const int b     = blockIdx.z;
    const int nBase = blockIdx.x * 256;
    const int oBase = blockIdx.y * 64;

    // Stage weight tile once: 64 rows x 256 halfs (row-major, contiguous)
    {
        int row = t >> 2, ck = (t & 3) * 64;
        const unsigned short* src = Wh + (size_t)(oBase + row) * C_ + ck;
        unsigned short* dst = &Wlds[row * WL_P + ck];
        #pragma unroll
        for (int j = 0; j < 8; ++j)
            *(v4u*)(dst + j * 8) = *(const v4u*)(src + j * 8);
    }

    const int wave = t >> 5, lane = t & 31, lr = lane & 15, hi = lane >> 4;
    const int kpair = t >> 4;          // 0..15 -> k rows 2kp, 2kp+1
    const int ngrp  = t & 15;          // 16 n columns each

    v8f acc[2][4];
    #pragma unroll
    for (int i = 0; i < 2; ++i)
        #pragma unroll
        for (int j = 0; j < 4; ++j) acc[i][j] = (v8f){};

    for (int k0 = 0; k0 < C_; k0 += 32) {
        __syncthreads();
        // Stage transposed X tile: Xt[n][k] f16, n=256, k=32
        {
            int c0 = k0 + 2 * kpair;
            int n  = nBase + ngrp * 16;
            unsigned pk[16];
            if (IN_F16) {
                const unsigned short* x =
                    (const unsigned short*)Xv + ((size_t)b * C_ + c0) * HW + n;
                if (k0 + 32 < C_) __builtin_prefetch(x + (size_t)32 * HW, 0, 1);
                v8us r0a = *(const v8us*)x,        r0b = *(const v8us*)(x + 8);
                v8us r1a = *(const v8us*)(x + HW), r1b = *(const v8us*)(x + HW + 8);
                #pragma unroll
                for (int j = 0; j < 8; ++j) {
                    pk[j]     = (unsigned)r0a[j] | ((unsigned)r1a[j] << 16);
                    pk[8 + j] = (unsigned)r0b[j] | ((unsigned)r1b[j] << 16);
                }
            } else {
                const float* x = (const float*)Xv + ((size_t)b * C_ + c0) * HW + n;
                if (k0 + 32 < C_) __builtin_prefetch(x + (size_t)32 * HW, 0, 1);
                v4f a0 = *(const v4f*)x,            a1 = *(const v4f*)(x + 4);
                v4f a2 = *(const v4f*)(x + 8),      a3 = *(const v4f*)(x + 12);
                v4f b0 = *(const v4f*)(x + HW),     b1 = *(const v4f*)(x + HW + 4);
                v4f b2 = *(const v4f*)(x + HW + 8), b3 = *(const v4f*)(x + HW + 12);
                #pragma unroll
                for (int j = 0; j < 4; ++j) {
                    pk[j]      = (unsigned)f2h(a0[j]) | ((unsigned)f2h(b0[j]) << 16);
                    pk[4 + j]  = (unsigned)f2h(a1[j]) | ((unsigned)f2h(b1[j]) << 16);
                    pk[8 + j]  = (unsigned)f2h(a2[j]) | ((unsigned)f2h(b2[j]) << 16);
                    pk[12 + j] = (unsigned)f2h(a3[j]) | ((unsigned)f2h(b3[j]) << 16);
                }
            }
            // Lane-rotated store order: 16 lanes hit 16 distinct banks.
            #pragma unroll
            for (int jj = 0; jj < 16; ++jj) {
                int j = (jj + ngrp) & 15;
                *(unsigned*)&Xt[(ngrp * 16 + j) * XT_P + 2 * kpair] = pk[j];
            }
        }
        __syncthreads();

        // A fragments: two 16-row n-tiles (lane lr row; K {0-7,16-23}/{8-15,24-31})
        ABfrag a[2];
        #pragma unroll
        for (int i = 0; i < 2; ++i) {
            const unsigned short* ar =
                &Xt[(wave * 32 + i * 16 + lr) * XT_P + hi * 8];
            a[i].h[0] = *(const v8h*)(ar);
            a[i].h[1] = *(const v8h*)(ar + 16);
        }
        #pragma unroll
        for (int j = 0; j < 4; ++j) {
            // B fragment: col = o, lane lr col; K {0-7,16-23}/{8-15,24-31}
            ABfrag bb;
            const unsigned short* br = &Wlds[(j * 16 + lr) * WL_P + k0 + hi * 8];
            bb.h[0] = *(const v8h*)(br);
            bb.h[1] = *(const v8h*)(br + 16);
            #pragma unroll
            for (int i = 0; i < 2; ++i)
                acc[i][j] = wmma16(a[i].v, bb.v, acc[i][j]);
        }
    }

    // Epilogue. Lane holds rows n = nRow..nRow+7 (consecutive), col o per tile.
    #pragma unroll
    for (int i = 0; i < 2; ++i) {
        const int nRow = nBase + wave * 32 + i * 16 + hi * 8;
        float sc[8];
        if (MODE != 2) {
            const float* sp = scale + (size_t)b * HW + nRow;
            v4f s0 = *(const v4f*)sp, s1 = *(const v4f*)(sp + 4);
            #pragma unroll
            for (int r = 0; r < 4; ++r) { sc[r] = s0[r]; sc[4 + r] = s1[r]; }
        }
        #pragma unroll
        for (int j = 0; j < 4; ++j) {
            const int o = oBase + j * 16 + lr;
            const float bo = bias[o];
            if (MODE == 2) {
                float* yp = (float*)Yv + ((size_t)b * OUTC + o) * HW + nRow;
                v4f o0, o1;
                #pragma unroll
                for (int r = 0; r < 4; ++r) {
                    o0[r] = acc[i][j][r] + bo;
                    o1[r] = acc[i][j][4 + r] + bo;
                }
                *(v4f*)yp = o0;
                *(v4f*)(yp + 4) = o1;
            } else {
                float pv[8];
                if (MODE == 0) {
                    const float* pp = pe + (size_t)o * HW + nRow;
                    v4f p0 = *(const v4f*)pp, p1 = *(const v4f*)(pp + 4);
                    #pragma unroll
                    for (int r = 0; r < 4; ++r) { pv[r] = p0[r]; pv[4 + r] = p1[r]; }
                } else {
                    #pragma unroll
                    for (int r = 0; r < 8; ++r) pv[r] = 0.0f;
                }
                union { unsigned short us[8]; v4u u4; } pko;
                #pragma unroll
                for (int r = 0; r < 8; ++r)
                    pko.us[r] = f2h(acc[i][j][r] * sc[r] + bo + pv[r]);
                unsigned short* yp =
                    (unsigned short*)Yv + ((size_t)b * C_ + o) * HW + nRow;
                *(v4u*)yp = pko.u4;
            }
        }
    }
}

// ---------------------------------------------------------------------------
// Per-(b,c) attention: logits = Q K^T (96x96x96), register softmax,
// O^T = V^T attn^T so everything stays contiguous. One block per slice,
// 192 thr = 6 waves, wave = 16-row tile. All matmuls WMMA f16.
// ---------------------------------------------------------------------------
constexpr int AP = 104;  // LDS pitch in halfs (96 + 8 pad)

__global__ void __launch_bounds__(192)
attn_kernel(const unsigned short* __restrict__ qH, const unsigned short* __restrict__ kH,
            const unsigned short* __restrict__ vH, unsigned short* __restrict__ outH)
{
    __shared__ unsigned short sQ[96 * AP];   // -> attn after softmax (wave-private rows)
    __shared__ unsigned short sK[96 * AP];
    __shared__ unsigned short sVt[96 * AP];  // V transposed: sVt[w][g]

    const int t = threadIdx.x;
    const size_t slice = (size_t)blockIdx.x * HW;   // gridDim.x = B*C
    const unsigned short* q = qH + slice;
    const unsigned short* k = kH + slice;
    const unsigned short* v = vH + slice;

    // Load Q, K row-major (48 halfs per thread each)
    {
        int row = t >> 1, half = t & 1;
        const unsigned short* qs = q + row * WW + half * 48;
        const unsigned short* ks = k + row * WW + half * 48;
        unsigned short* qd = &sQ[row * AP + half * 48];
        unsigned short* kd = &sK[row * AP + half * 48];
        #pragma unroll
        for (int j = 0; j < 6; ++j) {
            *(v4u*)(qd + j * 8) = *(const v4u*)(qs + j * 8);
            *(v4u*)(kd + j * 8) = *(const v4u*)(ks + j * 8);
        }
    }
    // Load V transposed (48 g-pairs x 12 w-groups = 576 tasks, 3 per thread)
    #pragma unroll
    for (int it = 0; it < 3; ++it) {
        int task = t + 192 * it;
        int g0 = (task / 12) * 2;
        int w0 = (task % 12) * 8;
        const unsigned short* vs = v + g0 * WW + w0;
        v8us r0 = *(const v8us*)vs;
        v8us r1 = *(const v8us*)(vs + WW);
        #pragma unroll
        for (int j = 0; j < 8; ++j)
            *(unsigned*)&sVt[(w0 + j) * AP + g0] =
                (unsigned)r0[j] | ((unsigned)r1[j] << 16);
    }
    __syncthreads();

    const int wave = t >> 5, lane = t & 31, lr = lane & 15, hi = lane >> 4;

    // logits[h, g] for h in wave's 16-row tile
    v8f acc[6];
    #pragma unroll
    for (int j = 0; j < 6; ++j) acc[j] = (v8f){};
    #pragma unroll
    for (int kt = 0; kt < 3; ++kt) {
        const int k0 = kt * 32;
        ABfrag a;
        const unsigned short* ar = &sQ[(wave * 16 + lr) * AP + k0 + hi * 8];
        a.h[0] = *(const v8h*)(ar);
        a.h[1] = *(const v8h*)(ar + 16);
        #pragma unroll
        for (int j = 0; j < 6; ++j) {
            ABfrag bb;   // B = K^T: col g -> contiguous K row
            const unsigned short* br = &sK[(j * 16 + lr) * AP + k0 + hi * 8];
            bb.h[0] = *(const v8h*)(br);
            bb.h[1] = *(const v8h*)(br + 16);
            acc[j] = wmma16(a.v, bb.v, acc[j]);
        }
    }

    // Register softmax: row h = wave*16 + r + 8*hi lives in VGPR r across the
    // 16-lane half-group (cols = j*16+lr). 4 shfl_xor steps per reduction.
    float inv[8];
    #pragma unroll
    for (int r = 0; r < 8; ++r) {
        float m = acc[0][r];
        #pragma unroll
        for (int j = 1; j < 6; ++j) m = fmaxf(m, acc[j][r]);
        #pragma unroll
        for (int msk = 1; msk < 16; msk <<= 1) m = fmaxf(m, __shfl_xor(m, msk, 32));
        float s = 0.0f;
        #pragma unroll
        for (int j = 0; j < 6; ++j) { acc[j][r] = __expf(acc[j][r] - m); s += acc[j][r]; }
        #pragma unroll
        for (int msk = 1; msk < 16; msk <<= 1) s += __shfl_xor(s, msk, 32);
        inv[r] = 1.0f / s;
    }
    // attn (f16) overwrites this wave's own rows of sQ
    #pragma unroll
    for (int j = 0; j < 6; ++j)
        #pragma unroll
        for (int r = 0; r < 8; ++r)
            sQ[(wave * 16 + r + 8 * hi) * AP + j * 16 + lr] = f2h(acc[j][r] * inv[r]);
    __syncthreads();

    // O^T[w, h] = sum_g Vt[w,g] * attn[h,g]   (A = Vt rows, B cols contiguous)
    v8f oacc[6];
    #pragma unroll
    for (int j = 0; j < 6; ++j) oacc[j] = (v8f){};
    #pragma unroll
    for (int kt = 0; kt < 3; ++kt) {
        const int k0 = kt * 32;
        ABfrag a;
        const unsigned short* ar = &sVt[(wave * 16 + lr) * AP + k0 + hi * 8];
        a.h[0] = *(const v8h*)(ar);
        a.h[1] = *(const v8h*)(ar + 16);
        #pragma unroll
        for (int j = 0; j < 6; ++j) {
            ABfrag bb;
            const unsigned short* br = &sQ[(j * 16 + lr) * AP + k0 + hi * 8];
            bb.h[0] = *(const v8h*)(br);
            bb.h[1] = *(const v8h*)(br + 16);
            oacc[j] = wmma16(a.v, bb.v, oacc[j]);
        }
    }
    // Store fused f16: lane holds 8 consecutive w for fixed h -> one b128 each
    unsigned short* op = outH + slice;
    #pragma unroll
    for (int j = 0; j < 6; ++j) {
        const int h  = j * 16 + lr;
        const int w0 = wave * 16 + hi * 8;
        union { unsigned short us[8]; v4u u4; } pk;
        #pragma unroll
        for (int r = 0; r < 8; ++r) pk.us[r] = f2h(oacc[j][r]);
        *(v4u*)(op + h * WW + w0) = pk.u4;
    }
}

// ---------------------------------------------------------------------------
extern "C" void kernel_launch(void* const* d_in, const int* in_sizes, int n_in,
                              void* d_out, int out_size, void* d_ws, size_t ws_size,
                              hipStream_t stream)
{
    const float* x1   = (const float*)d_in[0];
    const float* x2   = (const float*)d_in[1];
    const float* wvis = (const float*)d_in[2];
    const float* wir  = (const float*)d_in[3];
    const float* q_w  = (const float*)d_in[4];
    const float* q_b  = (const float*)d_in[5];
    const float* k_w  = (const float*)d_in[6];
    const float* k_b  = (const float*)d_in[7];
    const float* v_w  = (const float*)d_in[8];
    const float* v_b  = (const float*)d_in[9];
    const float* p_w  = (const float*)d_in[10];
    const float* p_b  = (const float*)d_in[11];

    char* ws = (char*)d_ws;
    float*          pe = (float*)(ws + OFF_PE);
    unsigned short* Wh = (unsigned short*)(ws + OFF_WH);
    unsigned short* qH = (unsigned short*)(ws + OFF_Q);
    unsigned short* kH = (unsigned short*)(ws + OFF_K);
    unsigned short* vH = (unsigned short*)(ws + OFF_V);
    unsigned short* fH = (unsigned short*)(ws + OFF_F);

    pe_kernel<<<dim3(C_ * HW / 256), dim3(256), 0, stream>>>(pe);
    wcvt_kernel<<<dim3((unsigned)(NWH / 256)), dim3(256), 0, stream>>>(q_w, k_w, v_w, p_w, Wh);

    const dim3 gG(HW / 256, C_ / 64, BATCH);   // 36 x 4 x 16
    const dim3 bG(256);
    const dim3 gA(BATCH * C_);                 // 4096 slices
    const dim3 bA(192);

    for (int s = 0; s < 3; ++s) {
        const unsigned short* WhQ = Wh + (size_t)s * CC;
        const unsigned short* WhK = Wh + (size_t)(3 + s) * CC;
        const unsigned short* WhV = Wh + (size_t)(6 + s) * CC;
        if (s == 0) {
            conv_gemm<false, 0><<<gG, bG, 0, stream>>>(x1, WhQ, q_b + s * C_, wvis, pe, qH);
            conv_gemm<false, 0><<<gG, bG, 0, stream>>>(x2, WhK, k_b + s * C_, wir,  pe, kH);
            conv_gemm<false, 1><<<gG, bG, 0, stream>>>(x2, WhV, v_b + s * C_, wir,  nullptr, vH);
        } else {
            conv_gemm<true, 0><<<gG, bG, 0, stream>>>(fH, WhQ, q_b + s * C_, wvis, pe, qH);
            conv_gemm<true, 0><<<gG, bG, 0, stream>>>(fH, WhK, k_b + s * C_, wir,  pe, kH);
            conv_gemm<true, 1><<<gG, bG, 0, stream>>>(fH, WhV, v_b + s * C_, wir,  nullptr, vH);
        }
        attn_kernel<<<gA, bA, 0, stream>>>(qH, kH, vH, fH);
    }

    // Projection: [B,64,HW] fp32 straight to d_out
    const dim3 gP(HW / 256, 1, BATCH);
    conv_gemm<true, 2><<<gP, bG, 0, stream>>>(fH, Wh + (size_t)9 * CC, p_b,
                                              nullptr, nullptr, d_out);
}
